// Attention_58987080843787
// MI455X (gfx1250) — matrix operand order repaired and checked by reading, not compiled, not run on hardware
//
#include <hip/hip_runtime.h>
#include <hip/hip_bf16.h>

// Problem dims (fixed by the reference)
#define BB   64
#define TT   2048
#define EE   512    // ENC_H == DEC_H
#define PP   256
#define CHUNK 64          // T-rows staged per block
#define NCH  (TT/CHUNK)   // 32 chunks per batch

typedef __attribute__((ext_vector_type(16))) _Float16 v16h;
typedef __attribute__((ext_vector_type(8)))  float    v8f;

// ---------------------------------------------------------------------------
// Generic M=64 GEMM: C[M,N] = A[M,K] @ B[K,N] (+bias), one wave per 16x16 tile.
// BTRANS==0: B[k][n] = Bm[k*ldb + n];  BTRANS==1: B[k][n] = Bm[n*ldb + k].
// Uses v_wmma_f32_16x16x32_f16 (f16 operands, f32 accumulate).
// Compile-time BTRANS/HASBIAS -> branch-free inner loop.
// ---------------------------------------------------------------------------
template <int BTRANS, int HASBIAS>
__global__ __launch_bounds__(32) void wmma_gemm64(
    const float* __restrict__ A, int lda,
    const float* __restrict__ Bm, int ldb,
    const float* __restrict__ bias,
    float* __restrict__ C, int ldc, int K) {
  const int nt = blockIdx.x;
  const int mt = blockIdx.y;
  const int l  = threadIdx.x;     // 0..31 (wave32)
  const int lh = l & 15;
  const int hi = (l >> 4) & 1;    // half-wave select
  const int rowA = mt * 16 + lh;  // A-matrix row for this lane
  const int colB = nt * 16 + lh;  // B-matrix col for this lane
  const int kbA = hi * 16;        // A K-base (contiguous k order, matches B)
  const int kbB = hi * 16;        // B K-base (contiguous k order, matches A)

  const float* __restrict__ Arow = A + (size_t)rowA * lda;
  const float* __restrict__ Brow = BTRANS ? (Bm + (size_t)colB * ldb) : Bm;

  v8f acc = {};
#pragma unroll 2
  for (int k0 = 0; k0 < K; k0 += 32) {
    v16h a, b;
    // A halves 0..7  <- K = k0+kbA+0..7   (contiguous, float4 x2)
    // A halves 8..15 <- K = k0+kbA+8..15  (contiguous, float4 x2)
    {
      const float4* p0 = (const float4*)(Arow + k0 + kbA);
      const float4* p1 = (const float4*)(Arow + k0 + kbA + 8);
      float4 a0 = p0[0], a1 = p0[1], a2 = p1[0], a3 = p1[1];
      a[0]=(_Float16)a0.x;  a[1]=(_Float16)a0.y;  a[2]=(_Float16)a0.z;  a[3]=(_Float16)a0.w;
      a[4]=(_Float16)a1.x;  a[5]=(_Float16)a1.y;  a[6]=(_Float16)a1.z;  a[7]=(_Float16)a1.w;
      a[8]=(_Float16)a2.x;  a[9]=(_Float16)a2.y;  a[10]=(_Float16)a2.z; a[11]=(_Float16)a2.w;
      a[12]=(_Float16)a3.x; a[13]=(_Float16)a3.y; a[14]=(_Float16)a3.z; a[15]=(_Float16)a3.w;
    }
    if (BTRANS) {
      // halves j <- Brow[k0+kbB+j], 16 contiguous floats (float4 x4)
      const float4* p = (const float4*)(Brow + k0 + kbB);
      float4 b0 = p[0], b1 = p[1], b2 = p[2], b3 = p[3];
      b[0]=(_Float16)b0.x;  b[1]=(_Float16)b0.y;  b[2]=(_Float16)b0.z;  b[3]=(_Float16)b0.w;
      b[4]=(_Float16)b1.x;  b[5]=(_Float16)b1.y;  b[6]=(_Float16)b1.z;  b[7]=(_Float16)b1.w;
      b[8]=(_Float16)b2.x;  b[9]=(_Float16)b2.y;  b[10]=(_Float16)b2.z; b[11]=(_Float16)b2.w;
      b[12]=(_Float16)b3.x; b[13]=(_Float16)b3.y; b[14]=(_Float16)b3.z; b[15]=(_Float16)b3.w;
    } else {
#pragma unroll
      for (int j = 0; j < 16; ++j)
        b[j] = (_Float16)Bm[(size_t)(k0 + kbB + j) * ldb + colB];
    }
    acc = __builtin_amdgcn_wmma_f32_16x16x32_f16(
        /*neg_a=*/false, a, /*neg_b=*/false, b,
        /*c_mod=*/(short)0, acc, /*reuse_a=*/false, /*reuse_b=*/false);
  }
#pragma unroll
  for (int i = 0; i < 8; ++i) {
    const int row = mt * 16 + i + hi * 8;   // C/D layout: VGPR i, half-wave hi
    const int col = nt * 16 + lh;
    float r = acc[i];
    if (HASBIAS) r += bias[col];
    C[(size_t)row * ldc + col] = r;
  }
}

// c[b] = bk . q[b]  (tiny)
__global__ __launch_bounds__(64) void bias_dot(
    const float* __restrict__ bk, const float* __restrict__ q,
    float* __restrict__ cb) {
  const int b = threadIdx.x;  // 64 threads, 1 block
  float s = 0.f;
  for (int p = 0; p < PP; ++p) s += bk[p] * q[b * PP + p];
  cb[b] = s;
}

// ---------------------------------------------------------------------------
// Single streaming pass over X (the only kernel that touches the 256 MB
// encoder_outputs). Stage a 64x512 tile into LDS with CDNA5 async-to-LDS
// copies (global_load_async_to_lds_b128, ASYNCcnt-tracked), compute logits
// s = X.kq + c, then per-block online-softmax partials (m, l, a[512]).
// ---------------------------------------------------------------------------
__global__ __launch_bounds__(256) void attn_stream(
    const float* __restrict__ X, const float* __restrict__ kq,
    const float* __restrict__ cb,
    float* __restrict__ s_buf,
    float* __restrict__ part_m, float* __restrict__ part_l,
    float* __restrict__ part_a) {
  extern __shared__ float sm[];
  float* xs   = sm;                 // CHUNK*EE  (staged X tile)
  float* kqs  = xs + CHUNK * EE;    // EE
  float* srow = kqs + EE;           // CHUNK + 1

  const int b  = blockIdx.y;
  const int ch = blockIdx.x;
  const int t0 = ch * CHUNK;
  const int tid = threadIdx.x;

  // ---- Async stage X[b, t0:t0+CHUNK, :] into LDS (16B granules, coalesced).
  // LDS byte address = low 32 bits of the generic pointer (ISA: addr[31:0]).
  {
    const float* gbase = X + ((size_t)b * TT + t0) * EE;
    const unsigned lbase = (unsigned)(uintptr_t)xs;
#pragma unroll
    for (int i = 0; i < (CHUNK * EE / 4) / 256; ++i) {   // 32 x b128 per thread
      const int e4 = tid + 256 * i;                      // float4 index
      const unsigned laddr = lbase + (unsigned)e4 * 16u;
      const unsigned long long gaddr =
          (unsigned long long)(uintptr_t)(gbase + (size_t)e4 * 4);
      asm volatile("global_load_async_to_lds_b128 %0, %1, off"
                   :: "v"(laddr), "v"(gaddr) : "memory");
    }
  }
  // Stage kq[b] with normal loads (needed immediately, DScnt-tracked stores)
  kqs[tid]       = kq[b * EE + tid];
  kqs[tid + 256] = kq[b * EE + tid + 256];
  // Drain ASYNCcnt before the workgroup barrier.
  asm volatile("s_wait_asynccnt 0x0" ::: "memory");
  __syncthreads();

  // Logits: 8 waves x 8 rows, 16 elems/lane, wave32 shuffle reduce
  const int wave = tid >> 5, lane = tid & 31;
  const float cbb = cb[b];
  for (int r = wave; r < CHUNK; r += 8) {
    float p = 0.f;
    const float* xr = xs + r * EE + lane * 16;
    const float* kr = kqs + lane * 16;
#pragma unroll
    for (int e = 0; e < 16; ++e) p += xr[e] * kr[e];
    for (int off = 16; off > 0; off >>= 1) p += __shfl_xor(p, off, 32);
    if (lane == 0) srow[r] = p + cbb;
  }
  __syncthreads();

  // Local max
  if (tid == 0) {
    float m = srow[0];
    for (int r = 1; r < CHUNK; ++r) m = fmaxf(m, srow[r]);
    srow[CHUNK] = m;
  }
  __syncthreads();
  const float ml = srow[CHUNK];

  // Save raw logits for the final attn write; replace with exp(s - ml)
  if (tid < CHUNK) {
    const float s = srow[tid];
    s_buf[(size_t)b * TT + t0 + tid] = s;
    srow[tid] = __expf(s - ml);
  }
  __syncthreads();

  if (tid == 0) {
    float lsum = 0.f;
    for (int r = 0; r < CHUNK; ++r) lsum += srow[r];
    part_m[b * NCH + ch] = ml;
    part_l[b * NCH + ch] = lsum;
  }

  // Partial weighted row-sum: a[e] = sum_r exp(s_r - ml) * xs[r][e]
  float* pa = part_a + ((size_t)(b * NCH + ch)) * EE;
  for (int e = tid; e < EE; e += 256) {
    float acc = 0.f;
#pragma unroll 4
    for (int r = 0; r < CHUNK; ++r) acc += srow[r] * xs[r * EE + e];
    pa[e] = acc;
  }
}

// Combine the 32 per-chunk partials of each batch: final M, l, ctxe = attn.X
__global__ __launch_bounds__(256) void attn_reduce(
    const float* __restrict__ part_m, const float* __restrict__ part_l,
    const float* __restrict__ part_a,
    float* __restrict__ Mb, float* __restrict__ lb,
    float* __restrict__ ctxe) {
  __shared__ float scale[NCH];
  __shared__ float lsh;
  const int b = blockIdx.x, tid = threadIdx.x;
  if (tid < 32) {
    const float m = part_m[b * NCH + tid];
    float mm = m;
    for (int off = 16; off > 0; off >>= 1) mm = fmaxf(mm, __shfl_xor(mm, off, 32));
    const float sc = __expf(m - mm);
    scale[tid] = sc;
    float li = sc * part_l[b * NCH + tid];
    for (int off = 16; off > 0; off >>= 1) li += __shfl_xor(li, off, 32);
    if (tid == 0) { lsh = li; Mb[b] = mm; lb[b] = li; }
  }
  __syncthreads();
  const float inv = 1.0f / lsh;
  for (int e = tid; e < EE; e += 256) {
    float acc = 0.f;
#pragma unroll 4
    for (int i = 0; i < NCH; ++i)
      acc += scale[i] * part_a[((size_t)(b * NCH + i)) * EE + e];
    ctxe[b * EE + e] = acc * inv;   // normalized: ctxe = attn . X
  }
}

// attn[b,t] = exp(s - M[b]) / l[b]
__global__ __launch_bounds__(256) void attn_write(
    const float* __restrict__ s_buf, const float* __restrict__ Mb,
    const float* __restrict__ lb, float* __restrict__ attn) {
  const int idx = blockIdx.x * 256 + threadIdx.x;   // 0 .. B*T-1
  const int b = idx / TT;
  attn[idx] = __expf(s_buf[idx] - Mb[b]) / lb[b];
}

extern "C" void kernel_launch(void* const* d_in, const int* in_sizes, int n_in,
                              void* d_out, int out_size, void* d_ws, size_t ws_size,
                              hipStream_t stream) {
  (void)in_sizes; (void)n_in; (void)out_size; (void)ws_size;
  const float* dec = (const float*)d_in[0];  // [64,512]
  const float* X   = (const float*)d_in[1];  // [64,2048,512]
  // d_in[2] = encoder_lens: unused by the reference (no masking)
  const float* Wk  = (const float*)d_in[3];  // [512,256]
  const float* bk  = (const float*)d_in[4];  // [256]
  const float* Wv  = (const float*)d_in[5];  // [512,256]
  const float* bv  = (const float*)d_in[6];  // [256]
  const float* Wq  = (const float*)d_in[7];  // [512,256]
  const float* bq  = (const float*)d_in[8];  // [256]

  float* outC = (float*)d_out;          // context [64,256]
  float* outA = outC + BB * PP;         // attn    [64,2048]

  float* ws    = (float*)d_ws;
  float* q     = ws;                         // 64*256
  float* kq    = q + BB * PP;                // 64*512
  float* cb    = kq + BB * EE;               // 64
  float* s_buf = cb + BB;                    // 64*2048
  float* pm    = s_buf + (size_t)BB * TT;    // 64*32
  float* pl    = pm + BB * NCH;              // 64*32
  float* pa    = pl + BB * NCH;              // 64*32*512
  float* Mb    = pa + (size_t)BB * NCH * EE; // 64
  float* lb    = Mb + BB;                    // 64
  float* ctxe  = lb + BB;                    // 64*512

  // q = dec @ Wq + bq          (M=64, N=256, K=512)
  wmma_gemm64<0, 1><<<dim3(PP / 16, BB / 16), 32, 0, stream>>>(
      dec, EE, Wq, PP, bq, q, PP, EE);
  // kq = q @ Wk^T              (M=64, N=512, K=256)  kq[b,e] = sum_p q[b,p]*Wk[e,p]
  wmma_gemm64<1, 0><<<dim3(EE / 16, BB / 16), 32, 0, stream>>>(
      q, PP, Wk, PP, nullptr, kq, EE, PP);
  // c[b] = bk . q[b]
  bias_dot<<<1, BB, 0, stream>>>(bk, q, cb);

  // One streaming pass over X (256 MB): logits + online softmax partials
  const size_t smem = (CHUNK * EE + EE + CHUNK + 8) * sizeof(float); // ~130.5 KB
  attn_stream<<<dim3(NCH, BB), 256, smem, stream>>>(X, kq, cb, s_buf, pm, pl, pa);

  // Cross-chunk combine -> M, l, ctxe (= attn . X)
  attn_reduce<<<BB, 256, 0, stream>>>(pm, pl, pa, Mb, lb, ctxe);

  // attn output
  attn_write<<<(BB * TT) / 256, 256, 0, stream>>>(s_buf, Mb, lb, outA);

  // context = ctxe @ Wv + bv   (M=64, N=256, K=512)
  wmma_gemm64<0, 1><<<dim3(PP / 16, BB / 16), 32, 0, stream>>>(
      ctxe, EE, Wv, PP, bv, outC, PP, EE);
}
